// InteractionIndexerWithLearnedRPE_52364241272947
// MI455X (gfx1250) — matrix operand order repaired and checked by reading, not compiled, not run on hardware
//
#include <hip/hip_runtime.h>
#include <hip/hip_bf16.h>

typedef __attribute__((ext_vector_type(16))) __bf16 v16bf;
typedef __attribute__((ext_vector_type(8)))  __bf16 v8bf;
typedef __attribute__((ext_vector_type(4)))  __bf16 v4bf;
typedef __attribute__((ext_vector_type(8)))  float  v8f;

#define NQ   2048
#define NK   4096
#define DIMK 1024
#define PROJ 256          // 2*H*D_I
#define NHEADS 8          // 4 data + 4 gate
#define MAXREL 128
#define NBUCKET (2*MAXREL + 1)

// ---------------------------------------------------------------------------
// Wait helpers
// ---------------------------------------------------------------------------
__device__ __forceinline__ void wait_async_zero() {
#if __has_builtin(__builtin_amdgcn_s_wait_asynccnt)
  __builtin_amdgcn_s_wait_asynccnt(0);
#else
  asm volatile("s_wait_asynccnt 0x0" ::: "memory");
#endif
}

// Async global -> LDS copy of 16 bytes (GLOBAL_LOAD_ASYNC_TO_LDS_B128, GV mode).
// lds_off: byte offset of destination within this workgroup's LDS.
__device__ __forceinline__ void async_copy_b128(unsigned int lds_off,
                                                unsigned long long gaddr) {
  asm volatile("global_load_async_to_lds_b128 %0, %1, off"
               :: "v"(lds_off), "v"(gaddr) : "memory");
}

// ---------------------------------------------------------------------------
// Fragment loaders (wave32, v_wmma_f32_16x16x32_bf16 operands, this code's contiguous K order).
//
// A (16x32, M x K): lane l holds row (l&15), K = (l>>4)*16 .. +15 contiguous.
// B (32x16, K x N): lane l holds column (l&15), K = (l>>4)*16 .. +15 contiguous.
// ---------------------------------------------------------------------------
__device__ __forceinline__ v16bf load_frag_a(const __bf16* __restrict__ base,
                                             int row0, int k0, int ld) {
  const int lane = threadIdx.x & 31;
  const __bf16* p = base + (size_t)(row0 + (lane & 15)) * ld + k0 + ((lane >> 4) << 4);
  v8bf lo = *(const v8bf*)p;          // K half .. +7
  v8bf hi = *(const v8bf*)(p + 8);    // K half+8 .. +15
  v16bf r;
#pragma unroll
  for (int i = 0; i < 8; ++i) { r[i] = lo[i]; r[i + 8] = hi[i]; }
  return r;
}

__device__ __forceinline__ v16bf load_frag_b(const __bf16* __restrict__ base,
                                             int col0, int k0, int ld) {
  const int lane = threadIdx.x & 31;
  const __bf16* p = base + (size_t)(col0 + (lane & 15)) * ld + k0 + ((lane >> 4) << 4);
  v8bf lo = *(const v8bf*)p;          // K half .. +7
  v8bf hi = *(const v8bf*)(p + 8);    // K half+8 .. +15
  v16bf r;
#pragma unroll
  for (int i = 0; i < 8; ++i) { r[i] = lo[i]; r[i + 8] = hi[i]; }
  return r;
}

// B fragment out of an LDS-resident [16][PROJ] bf16 strip (ds_load_b128 x2).
__device__ __forceinline__ v16bf load_frag_b_lds(const __bf16* s, int k0) {
  const int lane = threadIdx.x & 31;
  const __bf16* p = s + (lane & 15) * PROJ + k0 + ((lane >> 4) << 4);
  v8bf lo = *(const v8bf*)p;
  v8bf hi = *(const v8bf*)(p + 8);
  v16bf r;
#pragma unroll
  for (int i = 0; i < 8; ++i) { r[i] = lo[i]; r[i + 8] = hi[i]; }
  return r;
}

// ---------------------------------------------------------------------------
// Kernel 1: f32 -> bf16 conversion (vectorized x4)
// ---------------------------------------------------------------------------
__global__ __launch_bounds__(256)
void cvt_f32_bf16(const float* __restrict__ src, __bf16* __restrict__ dst, int n4) {
  int i = blockIdx.x * blockDim.x + threadIdx.x;
  if (i < n4) {
    float4 v = ((const float4*)src)[i];
    v4bf o;
    o[0] = (__bf16)v.x; o[1] = (__bf16)v.y; o[2] = (__bf16)v.z; o[3] = (__bf16)v.w;
    ((v4bf*)dst)[i] = o;
  }
}

// ---------------------------------------------------------------------------
// Kernel 2: C(N x 256) = X(N x 1024) * W^T(256 x 1024), bf16 in, bf16 out.
// One 16x64 strip per wave (A fragment reused across 4 WMMAs per K-step),
// 4 waves per block cover all 256 output columns. Grid: (N/16, 1).
// ---------------------------------------------------------------------------
__global__ __launch_bounds__(128)
void proj_gemm_bf16(const __bf16* __restrict__ X, const __bf16* __restrict__ W,
                    __bf16* __restrict__ C) {
  const int lane = threadIdx.x & 31;
  const int wave = threadIdx.x >> 5;
  const int m0 = blockIdx.x * 16;
  const int n0 = wave * 64;

  v8f c[4] = {};
#pragma unroll 2
  for (int k0 = 0; k0 < DIMK; k0 += 32) {
    v16bf a = load_frag_a(X, m0, k0, DIMK);
#pragma unroll
    for (int t = 0; t < 4; ++t) {
      v16bf b = load_frag_b(W, n0 + t * 16, k0, DIMK);
      c[t] = __builtin_amdgcn_wmma_f32_16x16x32_bf16(false, a, false, b,
                                                     (short)0, c[t], false, false);
    }
  }

  const int r0 = m0 + ((lane >> 4) << 3);
#pragma unroll
  for (int t = 0; t < 4; ++t) {
    const int col = n0 + t * 16 + (lane & 15);
#pragma unroll
    for (int j = 0; j < 8; ++j)
      C[(size_t)(r0 + j) * PROJ + col] = (__bf16)c[t][j];
  }
}

// ---------------------------------------------------------------------------
// Kernel 3: fused per-head dot (WMMA) + RPE + gate MLP + sigmoid + reduce.
// Block = 8 waves. Block tile: 128 q-rows x 64 k-cols.
//   - 64-col k strip (32 KB bf16) staged into LDS via async global->LDS copies
//   - RPE table (257x4 f32) staged into LDS
//   - wave w: q-rows [q0+16w, q0+16w+16), 8 A fragments loaded once,
//     4 k-subtiles x 8 WMMAs with B fragments served from LDS.
// Grid: (NK/64, NQ/128).
// ---------------------------------------------------------------------------
__global__ __launch_bounds__(256)
void fused_rpe_gate(const __bf16* __restrict__ qb, const __bf16* __restrict__ kb,
                    const int* __restrict__ qpos, const int* __restrict__ kpos,
                    const float* __restrict__ rpe_table,
                    const float* __restrict__ w1, const float* __restrict__ b1,
                    const float* __restrict__ w2, const float* __restrict__ b2,
                    float* __restrict__ out) {
  __shared__ __align__(16) __bf16 s_k[64 * PROJ];        // 32 KB k strip
  __shared__ __align__(16) float  s_rpe[NBUCKET * 4];    // 4 KB RPE table

  const int tid  = threadIdx.x;
  const int lane = tid & 31;
  const int wave = tid >> 5;
  const int k0   = blockIdx.x * 64;              // first k column of block
  const int q0   = blockIdx.y * 128 + wave * 16; // this wave's q rows

  // ---- stage k strip: linear 32 KB region kb[k0*PROJ ...] -> s_k (async) ----
  {
    const unsigned long long gbase =
        (unsigned long long)(uintptr_t)(kb + (size_t)k0 * PROJ);
    const unsigned int lbase = (unsigned int)(uintptr_t)(&s_k[0]);
#pragma unroll
    for (int r = 0; r < 8; ++r) {                 // 2048 x 16B chunks / 256 thr
      const unsigned int off = (unsigned int)(tid + r * 256) * 16u;
      async_copy_b128(lbase + off, gbase + off);
    }
  }
  // ---- stage RPE table (plain LDS stores overlap with async copies) ----
  for (int i = tid; i < NBUCKET * 4; i += 256) s_rpe[i] = rpe_table[i];

  // ---- A fragments (global, reused across all 4 subtiles) ----
  v16bf a[NHEADS];
#pragma unroll
  for (int h = 0; h < NHEADS; ++h) a[h] = load_frag_a(qb, q0, h * 32, PROJ);

  const int rbase = (lane >> 4) << 3;
  int qp[8];
#pragma unroll
  for (int j = 0; j < 8; ++j) qp[j] = qpos[q0 + rbase + j];

  wait_async_zero();
  __syncthreads();

  for (int sub = 0; sub < 4; ++sub) {
    const __bf16* s_sub = &s_k[sub * 16 * PROJ];

    v8f acc[NHEADS];
#pragma unroll
    for (int h = 0; h < NHEADS; ++h) {
      v16bf bf = load_frag_b_lds(s_sub, h * 32);
      v8f z = {};
      acc[h] = __builtin_amdgcn_wmma_f32_16x16x32_bf16(false, a[h], false, bf,
                                                       (short)0, z, false, false);
    }

    const int col = k0 + sub * 16 + (lane & 15);
    const int kpv = kpos[col];

#pragma unroll
    for (int j = 0; j < 8; ++j) {
      int rel = qp[j] - kpv;
      rel = rel < -MAXREL ? -MAXREL : (rel > MAXREL ? MAXREL : rel);
      const int bucket = rel + MAXREL;
      float4 rp = *(const float4*)&s_rpe[bucket * 4];

      float din[4], gin[4];
      din[0] = fmaxf(acc[0][j] + rp.x, 0.0f); gin[0] = acc[4][j] + rp.x;
      din[1] = fmaxf(acc[1][j] + rp.y, 0.0f); gin[1] = acc[5][j] + rp.y;
      din[2] = fmaxf(acc[2][j] + rp.z, 0.0f); gin[2] = acc[6][j] + rp.z;
      din[3] = fmaxf(acc[3][j] + rp.w, 0.0f); gin[3] = acc[7][j] + rp.w;

      float gate[4];
#pragma unroll
      for (int h = 0; h < 4; ++h) gate[h] = b2[h];          // uniform -> s_load
#pragma unroll
      for (int e = 0; e < 8; ++e) {
        float hid = b1[e];
#pragma unroll
        for (int h = 0; h < 4; ++h) hid = fmaf(gin[h], w1[e * 4 + h], hid);
        hid = fmaxf(hid, 0.0f);
#pragma unroll
        for (int h = 0; h < 4; ++h) gate[h] = fmaf(w2[h * 8 + e], hid, gate[h]);
      }

      float val = 0.0f;
#pragma unroll
      for (int h = 0; h < 4; ++h)
        val = fmaf(din[h], __builtin_amdgcn_rcpf(1.0f + __expf(-gate[h])), val);

      out[(size_t)(q0 + rbase + j) * NK + col] = val;
    }
  }
}

// ---------------------------------------------------------------------------
// Launch
// ---------------------------------------------------------------------------
extern "C" void kernel_launch(void* const* d_in, const int* in_sizes, int n_in,
                              void* d_out, int out_size, void* d_ws, size_t ws_size,
                              hipStream_t stream) {
  const float* query_states = (const float*)d_in[0];   // 2048x1024
  const float* key_states   = (const float*)d_in[1];   // 4096x1024
  const int*   query_pos    = (const int*)d_in[2];     // 2048
  const int*   key_pos      = (const int*)d_in[3];     // 4096
  const float* Wq           = (const float*)d_in[4];   // 256x1024
  const float* Wk           = (const float*)d_in[5];   // 256x1024
  const float* rpe_table    = (const float*)d_in[6];   // 257x4
  const float* gate_w1      = (const float*)d_in[7];   // 8x4
  const float* gate_b1      = (const float*)d_in[8];   // 8
  const float* gate_w2      = (const float*)d_in[9];   // 4x8
  const float* gate_b2      = (const float*)d_in[10];  // 4
  float* out = (float*)d_out;

  // Workspace layout (bf16), 16 MB total.
  uint8_t* w = (uint8_t*)d_ws;
  __bf16* qs_b = (__bf16*)(w + 0);                          // 2048x1024
  __bf16* ks_b = (__bf16*)(w + (size_t)NQ * DIMK * 2);      // 4096x1024
  __bf16* wq_b = (__bf16*)(w + (size_t)(NQ + NK) * DIMK * 2);
  __bf16* wk_b = (__bf16*)(w + (size_t)(NQ + NK + PROJ) * DIMK * 2);
  __bf16* q_b  = (__bf16*)(w + (size_t)(NQ + NK + 2 * PROJ) * DIMK * 2);       // 2048x256
  __bf16* k_b  = (__bf16*)(w + (size_t)(NQ + NK + 2 * PROJ) * DIMK * 2
                             + (size_t)NQ * PROJ * 2);                          // 4096x256

  // 1) f32 -> bf16 conversions
  {
    int n4;
    n4 = NQ * DIMK / 4;
    cvt_f32_bf16<<<(n4 + 255) / 256, 256, 0, stream>>>(query_states, qs_b, n4);
    n4 = NK * DIMK / 4;
    cvt_f32_bf16<<<(n4 + 255) / 256, 256, 0, stream>>>(key_states, ks_b, n4);
    n4 = PROJ * DIMK / 4;
    cvt_f32_bf16<<<(n4 + 255) / 256, 256, 0, stream>>>(Wq, wq_b, n4);
    cvt_f32_bf16<<<(n4 + 255) / 256, 256, 0, stream>>>(Wk, wk_b, n4);
  }

  // 2) projections: q = Xq * Wq^T, k = Xk * Wk^T  (bf16 WMMA, f32 accum)
  proj_gemm_bf16<<<dim3(NQ / 16, 1), 128, 0, stream>>>(qs_b, wq_b, q_b);
  proj_gemm_bf16<<<dim3(NK / 16, 1), 128, 0, stream>>>(ks_b, wk_b, k_b);

  // 3) fused per-head dot + RPE + gate MLP (k strip staged via async->LDS)
  fused_rpe_gate<<<dim3(NK / 64, NQ / 128), 256, 0, stream>>>(
      q_b, k_b, query_pos, key_pos, rpe_table,
      gate_w1, gate_b1, gate_w2, gate_b2, out);
}
